// MambaEncoder_10479720203035
// MI455X (gfx1250) — matrix operand order repaired and checked by reading, not compiled, not run on hardware
//
#include <hip/hip_runtime.h>
#include <cstdint>
#include <cstddef>

// ---------------- model dims ----------------
#define D_MODELC 1024
#define N_LAYERSC 4
#define FFN_DIM 3584
#define D_STATEC 16
#define D_CONVC 4
#define D_INNERC 2048
#define DT_RANKC 64
#define BATCHC 4
#define SEQC 2048
#define MTROWS (BATCHC * SEQC) /* 8192 */

typedef __attribute__((ext_vector_type(16))) __bf16 v16bf;
typedef __attribute__((ext_vector_type(8)))  float  v8f;

__device__ __forceinline__ __bf16 f2bf(float f) {
  unsigned int u = __builtin_bit_cast(unsigned int, f);
  unsigned int r = (u + 0x7fffu + ((u >> 16) & 1u)) >> 16; // RNE
  unsigned short h = (unsigned short)r;
  return __builtin_bit_cast(__bf16, h);
}

// LDS byte offset (addrspace(3) pointer value) for async-to-LDS destinations
__device__ __forceinline__ unsigned as_lds(const void* p) {
  return (unsigned)(unsigned long long)(__attribute__((address_space(3))) const char*)p;
}

// CDNA5 async copy: 16B per lane, memory -> LDS, tracked by ASYNCcnt
__device__ __forceinline__ void async_b128(unsigned lds_byte, const void* g) {
  unsigned long long ga = (unsigned long long)g;
  asm volatile("global_load_async_to_lds_b128 %0, %1, off"
               :: "v"(lds_byte), "v"(ga) : "memory");
}
__device__ __forceinline__ void wait_async0() {
  asm volatile("s_wait_asynccnt 0x0" ::: "memory");
}

// ---------------- elementwise / small kernels ----------------

__global__ __launch_bounds__(256) void cvt_bf16_kernel(const float* __restrict__ s,
                                                       __bf16* __restrict__ d, int n) {
  int i = blockIdx.x * 256 + threadIdx.x;
  if (i < n) d[i] = f2bf(s[i]);
}

// x + sinusoidal positional encoding
__global__ __launch_bounds__(256) void pe_kernel(const float* __restrict__ x,
                                                 float* __restrict__ o, int n) {
  int i = blockIdx.x * 256 + threadIdx.x;
  if (i >= n) return;
  int c = i & (D_MODELC - 1);
  int t = (i / D_MODELC) & (SEQC - 1);
  int j2 = c & ~1;  // 2*j
  float div = expf((float)j2 * (-9.210340371976184f / (float)D_MODELC)); // -ln(10000)/D
  float ang = (float)t * div;
  float pe = (c & 1) ? cosf(ang) : sinf(ang);
  o[i] = x[i] + pe;
}

// LayerNorm over D_MODEL=1024, one block per row; optional f32 and bf16 outputs
__global__ __launch_bounds__(256) void ln_kernel(const float* __restrict__ x,
                                                 const float* __restrict__ g,
                                                 const float* __restrict__ bta,
                                                 float* __restrict__ outf,
                                                 __bf16* __restrict__ outb) {
  __shared__ float red[256];
  const int tid = threadIdx.x;
  const size_t row = blockIdx.x;
  const float* xr = x + row * D_MODELC;
  float v[4];
  float s = 0.f;
#pragma unroll
  for (int j = 0; j < 4; ++j) { v[j] = xr[tid + j * 256]; s += v[j]; }
  red[tid] = s; __syncthreads();
  for (int off = 128; off > 0; off >>= 1) { if (tid < off) red[tid] += red[tid + off]; __syncthreads(); }
  float mu = red[0] * (1.f / D_MODELC);
  __syncthreads();
  s = 0.f;
#pragma unroll
  for (int j = 0; j < 4; ++j) { float d = v[j] - mu; s += d * d; }
  red[tid] = s; __syncthreads();
  for (int off = 128; off > 0; off >>= 1) { if (tid < off) red[tid] += red[tid + off]; __syncthreads(); }
  float rstd = rsqrtf(red[0] * (1.f / D_MODELC) + 1e-5f);
#pragma unroll
  for (int j = 0; j < 4; ++j) {
    int c = tid + j * 256;
    float o = (v[j] - mu) * rstd * g[c] + bta[c];
    if (outf) outf[row * D_MODELC + c] = o;
    if (outb) outb[row * D_MODELC + c] = f2bf(o);
  }
}

// causal depthwise conv (K=4) + bias + SiLU; reads x-half of xz [rows, 4096]
__global__ __launch_bounds__(256) void conv_silu_kernel(const float* __restrict__ xz,
                                                        const float* __restrict__ cw,
                                                        const float* __restrict__ cb,
                                                        float* __restrict__ xc,
                                                        __bf16* __restrict__ xcb, int n) {
  int i = blockIdx.x * 256 + threadIdx.x;
  if (i >= n) return;
  int d = i & (D_INNERC - 1);
  int row = i / D_INNERC;          // b*SEQ + t
  int t = row & (SEQC - 1);
  float acc = cb[d];
#pragma unroll
  for (int k = 0; k < D_CONVC; ++k) {
    int tt = t - (D_CONVC - 1) + k;
    if (tt >= 0)
      acc = fmaf(cw[d * D_CONVC + k],
                 xz[(size_t)(row - (D_CONVC - 1) + k) * (2 * D_INNERC) + d], acc);
  }
  float s = acc / (1.f + __expf(-acc)); // SiLU
  xc[i] = s;
  xcb[i] = f2bf(s);
}

// selective scan: one thread per (batch, channel); state in VGPRs
__global__ __launch_bounds__(256) void scan_kernel(const float* __restrict__ xc,
                                                   const float* __restrict__ dt,
                                                   const float* __restrict__ xd,
                                                   const float* __restrict__ xz,
                                                   const float* __restrict__ a_log,
                                                   const float* __restrict__ dskip,
                                                   __bf16* __restrict__ yb) {
  int gid = blockIdx.x * 256 + threadIdx.x;       // < BATCH*D_INNER
  int b = gid >> 11;                              // / 2048
  int d = gid & (D_INNERC - 1);
  float A[D_STATEC];
#pragma unroll
  for (int n = 0; n < D_STATEC; ++n) A[n] = -__expf(a_log[d * D_STATEC + n]);
  float dsk = dskip[d];
  float s[D_STATEC];
#pragma unroll
  for (int n = 0; n < D_STATEC; ++n) s[n] = 0.f;
  for (int t = 0; t < SEQC; ++t) {
    size_t row = (size_t)(b * SEQC + t);
    float xv = xc[row * D_INNERC + d];
    float dv = dt[row * D_INNERC + d];
    const float* Bp = xd + row * (DT_RANKC + 2 * D_STATEC) + DT_RANKC;
    const float* Cp = Bp + D_STATEC;
    float xdt = xv * dv;
    float y = 0.f;
#pragma unroll
    for (int n = 0; n < D_STATEC; ++n) {
      s[n] = fmaf(s[n], __expf(dv * A[n]), xdt * Bp[n]);
      y = fmaf(s[n], Cp[n], y);
    }
    y = fmaf(dsk, xv, y);
    float zv = xz[row * (2 * D_INNERC) + D_INNERC + d];
    y *= zv / (1.f + __expf(-zv)); // * silu(z)
    yb[row * D_INNERC + d] = f2bf(y);
  }
}

// length mask (also used to emit the final output)
__global__ __launch_bounds__(256) void mask_kernel(const float* __restrict__ s,
                                                   float* __restrict__ d,
                                                   const int* __restrict__ lengths, int n) {
  int i = blockIdx.x * 256 + threadIdx.x;
  if (i >= n) return;
  int row = i / D_MODELC;
  int t = row & (SEQC - 1);
  int b = row >> 11;
  d[i] = (t < lengths[b]) ? s[i] : 0.f;
}

// ---------------- WMMA GEMM: C[M,N] = act(A[M,K] * W[N,K]^T + bias) + resid ----------------
// 256 thr = 8 waves; block tile 64x256; wave tile 32x64 = 2x4 v_wmma_f32_16x16x32_bf16.
// Double-buffered LDS fed by global_load_async_to_lds_b128 (ASYNCcnt), overlapped with WMMA.
#define BMT 64
#define BNT 256
#define BKT 32
#define LROW 40 /* BKT+8: 80B row stride = 5x16B (aligned b128) and bank-conflict-free */

union Frag { v16bf v; uint32_t u[8]; uint4 q[2]; };

__device__ __forceinline__ float act_apply(float v, int act) {
  if (act == 1) return 0.5f * v * (1.f + erff(v * 0.70710678118654752f)); // exact GELU
  if (act == 2) return fmaxf(v, 0.f) + log1pf(__expf(-fabsf(v)));        // softplus
  return v;
}

__global__ __launch_bounds__(256) void gemm_bf16_kernel(
    const __bf16* __restrict__ A, int lda, const __bf16* __restrict__ W,
    const float* __restrict__ bias, const float* __restrict__ resid,
    float* __restrict__ Cf, __bf16* __restrict__ Cb,
    int M, int N, int K, int act) {
  (void)M;
  __shared__ __bf16 sA[2][BMT][LROW];
  __shared__ __bf16 sW[2][BNT][LROW];

  const int tid = threadIdx.x;
  const int lane = tid & 31;
  const int wave = tid >> 5;      // 0..7
  const int wm = wave >> 2;       // 0..1 (M)
  const int wn = wave & 3;        // 0..3 (N)
  const int half = lane >> 4;     // lane half per ISA layout
  const int l15 = lane & 15;
  const int bm = blockIdx.y * BMT;
  const int bn = blockIdx.x * BNT;

  // async-copy coordinates: A tile -> 1 b128/thread, W tile -> one row (4 b128)/thread
  const int ar = tid >> 2, ac = (tid & 3) * 8;
  int nrow = bn + tid;
  if (nrow > N - 1) nrow = N - 1;   // clamp: OOB cols never stored by epilogue
  const __bf16* Asrc = A + (size_t)(bm + ar) * lda + ac;
  const __bf16* Wsrc = W + (size_t)nrow * K;

  unsigned ldsA[2], ldsW[2];
  ldsA[0] = as_lds(&sA[0][ar][ac]);  ldsA[1] = as_lds(&sA[1][ar][ac]);
  ldsW[0] = as_lds(&sW[0][tid][0]);  ldsW[1] = as_lds(&sW[1][tid][0]);

  v8f acc[2][4] = {};
  const int nk = K / BKT;

  // prologue: stage 0 copies in flight
  async_b128(ldsA[0], Asrc);
#pragma unroll
  for (int j = 0; j < 4; ++j) async_b128(ldsW[0] + j * 16, Wsrc + j * 8);

  for (int k = 0; k < nk; ++k) {
    const int cur = k & 1;
    wait_async0();       // this wave's copies into buf[cur] are in LDS
    __syncthreads();     // everyone's copies visible

    if (k + 1 < nk) {    // kick stage k+1 into the other buffer while we compute
      const int kb = (k + 1) * BKT;
      async_b128(ldsA[cur ^ 1], Asrc + kb);
#pragma unroll
      for (int j = 0; j < 4; ++j) async_b128(ldsW[cur ^ 1] + j * 16, Wsrc + kb + j * 8);
    }

    // fragments per ISA 7.12.2: A lane->M, dwords {4*half}/{8+4*half}; B lane->N, dwords {4*half}/{8+4*half}
    Frag fa[2], fb[4];
#pragma unroll
    for (int i = 0; i < 2; ++i) {
      const uint32_t* ap = (const uint32_t*)(&sA[cur][wm * 32 + i * 16 + l15][0]);
      fa[i].q[0] = *(const uint4*)(ap + 4 * half);
      fa[i].q[1] = *(const uint4*)(ap + 8 + 4 * half);
    }
#pragma unroll
    for (int j = 0; j < 4; ++j) {
      const uint32_t* wp = (const uint32_t*)(&sW[cur][wn * 64 + j * 16 + l15][0]);
      fb[j].q[0] = *(const uint4*)(wp + 4 * half);
      fb[j].q[1] = *(const uint4*)(wp + 8 + 4 * half);
    }
#pragma unroll
    for (int fm = 0; fm < 2; ++fm)
#pragma unroll
      for (int fn = 0; fn < 4; ++fn)
        acc[fm][fn] = __builtin_amdgcn_wmma_f32_16x16x32_bf16(
            false, fa[fm].v, false, fb[fn].v, (short)0, acc[fm][fn], false, false);
    __syncthreads();     // all reads of buf[cur] done before it is overwritten
  }

  // epilogue: bias -> activation -> residual; f32 and/or bf16 stores
#pragma unroll
  for (int fm = 0; fm < 2; ++fm)
#pragma unroll
    for (int fn = 0; fn < 4; ++fn)
#pragma unroll
      for (int r = 0; r < 8; ++r) {
        int row = bm + wm * 32 + fm * 16 + half * 8 + r; // D layout: VGPR r -> M=r / M=r+8
        int col = bn + wn * 64 + fn * 16 + l15;
        if (col < N) {
          float v = acc[fm][fn][r];
          if (bias) v += bias[col];
          v = act_apply(v, act);
          size_t idx = (size_t)row * N + col;
          if (resid) v += resid[idx];
          if (Cf) Cf[idx] = v;
          if (Cb) Cb[idx] = f2bf(v);
        }
      }
}

// ---------------- host orchestration ----------------

extern "C" void kernel_launch(void* const* d_in, const int* in_sizes, int n_in,
                              void* d_out, int out_size, void* d_ws, size_t ws_size,
                              hipStream_t stream) {
  (void)in_sizes; (void)n_in; (void)out_size; (void)ws_size;

  const float* x_in    = (const float*)d_in[0];
  const int*   lengths = (const int*)d_in[1];
  const float* ln0_g   = (const float*)d_in[2];
  const float* ln0_b   = (const float*)d_in[3];
  const float* ln1_g   = (const float*)d_in[4];
  const float* ln1_b   = (const float*)d_in[5];
  const float* in_w    = (const float*)d_in[6];
  const float* conv_w  = (const float*)d_in[7];
  const float* conv_b  = (const float*)d_in[8];
  const float* xp_w    = (const float*)d_in[9];
  const float* dt_w    = (const float*)d_in[10];
  const float* dt_b    = (const float*)d_in[11];
  const float* A_log   = (const float*)d_in[12];
  const float* D_skip  = (const float*)d_in[13];
  const float* out_w   = (const float*)d_in[14];
  const float* ln2_g   = (const float*)d_in[15];
  const float* ln2_b   = (const float*)d_in[16];
  const float* f1_w    = (const float*)d_in[17];
  const float* f1_b    = (const float*)d_in[18];
  const float* f2_w    = (const float*)d_in[19];
  const float* f2_b    = (const float*)d_in[20];

  // workspace carve-out
  char* p = (char*)d_ws;
  auto alloc = [&](size_t bytes) -> void* {
    void* r = (void*)p;
    p += (bytes + 255) & ~(size_t)255;
    return r;
  };
  const size_t XD = DT_RANKC + 2 * D_STATEC; // 96
  __bf16* w_in  = (__bf16*)alloc((size_t)N_LAYERSC * 2 * D_INNERC * D_MODELC * 2);
  __bf16* w_xp  = (__bf16*)alloc((size_t)N_LAYERSC * XD * D_INNERC * 2);
  __bf16* w_dt  = (__bf16*)alloc((size_t)N_LAYERSC * D_INNERC * DT_RANKC * 2);
  __bf16* w_out = (__bf16*)alloc((size_t)N_LAYERSC * D_MODELC * D_INNERC * 2);
  __bf16* w_f1  = (__bf16*)alloc((size_t)N_LAYERSC * FFN_DIM * D_MODELC * 2);
  __bf16* w_f2  = (__bf16*)alloc((size_t)N_LAYERSC * D_MODELC * FFN_DIM * 2);
  float*  xbuf  = (float*) alloc((size_t)MTROWS * D_MODELC * 4);
  __bf16* hbf   = (__bf16*)alloc((size_t)MTROWS * D_MODELC * 2);
  float*  xz    = (float*) alloc((size_t)MTROWS * 2 * D_INNERC * 4);
  float*  xc    = (float*) alloc((size_t)MTROWS * D_INNERC * 4);
  __bf16* xcb   = (__bf16*)alloc((size_t)MTROWS * D_INNERC * 2);
  float*  xd    = (float*) alloc((size_t)MTROWS * XD * 4);
  __bf16* xdb   = (__bf16*)alloc((size_t)MTROWS * XD * 2);
  float*  dtbuf = (float*) alloc((size_t)MTROWS * D_INNERC * 4);
  __bf16* yb    = (__bf16*)alloc((size_t)MTROWS * D_INNERC * 2);
  __bf16* gb    = (__bf16*)alloc((size_t)MTROWS * FFN_DIM * 2);

  auto cvt = [&](const float* s, __bf16* d, size_t n) {
    cvt_bf16_kernel<<<dim3((unsigned)((n + 255) / 256)), dim3(256), 0, stream>>>(s, d, (int)n);
  };
  auto gemm = [&](const __bf16* A, int lda, const __bf16* W, const float* bias,
                  const float* res, float* Cf, __bf16* Cb, int M, int N, int K, int act) {
    dim3 g((unsigned)((N + BNT - 1) / BNT), (unsigned)(M / BMT));
    gemm_bf16_kernel<<<g, 256, 0, stream>>>(A, lda, W, bias, res, Cf, Cb, M, N, K, act);
  };

  // weights -> bf16 (resident in 192MB L2 across the pass)
  cvt(in_w,  w_in,  (size_t)N_LAYERSC * 2 * D_INNERC * D_MODELC);
  cvt(xp_w,  w_xp,  (size_t)N_LAYERSC * XD * D_INNERC);
  cvt(dt_w,  w_dt,  (size_t)N_LAYERSC * D_INNERC * DT_RANKC);
  cvt(out_w, w_out, (size_t)N_LAYERSC * D_MODELC * D_INNERC);
  cvt(f1_w,  w_f1,  (size_t)N_LAYERSC * FFN_DIM * D_MODELC);
  cvt(f2_w,  w_f2,  (size_t)N_LAYERSC * D_MODELC * FFN_DIM);

  // x = x + PE ; layer-0 pre-LN replaces x
  pe_kernel<<<(MTROWS * D_MODELC) / 256, 256, 0, stream>>>(x_in, xbuf, MTROWS * D_MODELC);
  ln_kernel<<<MTROWS, 256, 0, stream>>>(xbuf, ln0_g, ln0_b, xbuf, nullptr);

  for (int i = 0; i < N_LAYERSC; ++i) {
    const __bf16* Wi_in  = w_in  + (size_t)i * 2 * D_INNERC * D_MODELC;
    const __bf16* Wi_xp  = w_xp  + (size_t)i * XD * D_INNERC;
    const __bf16* Wi_dt  = w_dt  + (size_t)i * D_INNERC * DT_RANKC;
    const __bf16* Wi_out = w_out + (size_t)i * D_MODELC * D_INNERC;
    const __bf16* Wi_f1  = w_f1  + (size_t)i * FFN_DIM * D_MODELC;
    const __bf16* Wi_f2  = w_f2  + (size_t)i * D_MODELC * FFN_DIM;

    // h = LN(x, ln1)
    ln_kernel<<<MTROWS, 256, 0, stream>>>(xbuf, ln1_g + i * D_MODELC, ln1_b + i * D_MODELC,
                                          nullptr, hbf);
    // xz = h @ in_w^T                           [8192 x 4096]
    gemm(hbf, D_MODELC, Wi_in, nullptr, nullptr, xz, nullptr, MTROWS, 2 * D_INNERC, D_MODELC, 0);
    // xc = silu(causal_conv(xz[:, :2048]))
    conv_silu_kernel<<<(MTROWS * D_INNERC) / 256, 256, 0, stream>>>(
        xz, conv_w + (size_t)i * D_INNERC * D_CONVC, conv_b + (size_t)i * D_INNERC,
        xc, xcb, MTROWS * D_INNERC);
    // xd = xc @ xp_w^T                          [8192 x 96]
    gemm(xcb, D_INNERC, Wi_xp, nullptr, nullptr, xd, xdb, MTROWS, (int)XD, D_INNERC, 0);
    // dt = softplus(xd[:, :64] @ dt_w^T + dt_b) [8192 x 2048]
    gemm(xdb, (int)XD, Wi_dt, dt_b + (size_t)i * D_INNERC, nullptr, dtbuf, nullptr,
         MTROWS, D_INNERC, DT_RANKC, 2);
    // selective scan + D-skip + silu(z) gate -> yb (bf16)
    scan_kernel<<<(BATCHC * D_INNERC) / 256, 256, 0, stream>>>(
        xc, dtbuf, xd, xz, A_log + (size_t)i * D_INNERC * D_STATEC,
        D_skip + (size_t)i * D_INNERC, yb);
    // x = x + y @ out_w^T
    gemm(yb, D_INNERC, Wi_out, nullptr, xbuf, xbuf, nullptr, MTROWS, D_MODELC, D_INNERC, 0);
    // h = LN(x, ln2)
    ln_kernel<<<MTROWS, 256, 0, stream>>>(xbuf, ln2_g + i * D_MODELC, ln2_b + i * D_MODELC,
                                          nullptr, hbf);
    // g = gelu(h @ f1^T + b1) (bf16)            [8192 x 3584]
    gemm(hbf, D_MODELC, Wi_f1, f1_b + (size_t)i * FFN_DIM, nullptr, nullptr, gb,
         MTROWS, FFN_DIM, D_MODELC, 1);
    // x = x + g @ f2^T + b2
    gemm(gb, FFN_DIM, Wi_f2, f2_b + (size_t)i * D_MODELC, xbuf, xbuf, nullptr,
         MTROWS, D_MODELC, FFN_DIM, 0);
    // x *= mask ; last layer writes d_out
    float* mdst = (i == N_LAYERSC - 1) ? (float*)d_out : xbuf;
    mask_kernel<<<(MTROWS * D_MODELC) / 256, 256, 0, stream>>>(xbuf, mdst, lengths,
                                                               MTROWS * D_MODELC);
  }
}
